// RWKV_Tmix_x060_infctx_26895085208307
// MI455X (gfx1250) — matrix operand order repaired and checked by reading, not compiled, not run on hardware
//
#include <hip/hip_runtime.h>
#include <hip/hip_bf16.h>

// ---------------- problem constants ----------------
constexpr int B  = 4;
constexpr int T  = 2048;
constexpr int C  = 2048;
constexpr int H  = 32;
constexpr int N  = 64;
constexpr int BT = B * T;          // 8192 rows
constexpr float GN_EPS = 1e-5f * 64.0f;  // 1e-5 * head_size_divisor^2

// ---------------- vector types ----------------
typedef __attribute__((ext_vector_type(16))) __bf16 v16bf;
typedef __attribute__((ext_vector_type(8)))  __bf16 v8bf;
typedef __attribute__((ext_vector_type(8)))  float  v8f;

union FragU {
  struct { v8bf lo, hi; } p;
  v16bf v;
};

// ---------------- device math helpers ----------------
__device__ inline float fast_tanh(float x) {
  float ax = fabsf(x);
  float e  = __expf(-2.0f * ax);       // in (0,1], ->0 for large |x|
  float r  = (1.0f - e) / (1.0f + e);
  return copysignf(r, x);
}

// ---------------- WMMA helpers (gfx1250, wave32) ----------------
__device__ inline v8f wmma_bf16(v16bf a, v16bf b, v8f c) {
  // D = A(16x32) * B(32x16) + C ; emits v_wmma_f32_16x16x32_bf16
  return __builtin_amdgcn_wmma_f32_16x16x32_bf16(
      /*neg_a=*/false, a, /*neg_b=*/false, b,
      /*c_mod=*/(short)0, c, /*reuse_a=*/false, /*reuse_b=*/false);
}

// A-matrix 16x32 bf16 fragment, A row-major [*, lda].
// Lane L: M = L&15 ; Kbase = (L<16)?0:16 ; halves 0..7 -> K=Kb..Kb+7,
// halves 8..15 -> K=Kb+8..Kb+15  => two contiguous 16B loads.
__device__ inline v16bf load_a_frag(const __bf16* A, int lda, int row0, int k0) {
  int lane = threadIdx.x & 31;
  int m  = lane & 15;
  int kb = (lane < 16) ? 0 : 16;
  const __bf16* p = A + (size_t)(row0 + m) * lda + (k0 + kb);
  FragU u;
  u.p.lo = *(const v8bf*)p;          // K = kb .. kb+7
  u.p.hi = *(const v8bf*)(p + 8);    // K = kb+8 .. kb+15
  return u.v;
}

// B-matrix 32x16 bf16 fragment for C = A * Wn^T where Wn is row-major [*, ldw]
// and B(k,n) = Wn[n, k] (column n of B = row n of Wn, contiguous in k).
// Lane L: N = L&15 ; Kbase = (L<16)?0:16 ; halves e -> K = Kb + e.
// Entire fragment is one aligned 32-byte load (two b128s, no packing).
__device__ inline v16bf load_b_fragT(const __bf16* W, int ldw, int col0, int k0) {
  int lane = threadIdx.x & 31;
  int n  = lane & 15;
  int kb = (lane < 16) ? 0 : 16;
  const __bf16* p = W + (size_t)(col0 + n) * ldw + (k0 + kb);
  return *(const v16bf*)p;
}

// ---------------- kernels ----------------

__global__ void cvt_bf16_kernel(const float* __restrict__ s, __bf16* __restrict__ d, size_t n) {
  size_t i = (size_t)blockIdx.x * blockDim.x + threadIdx.x;
  if (i < n) d[i] = (__bf16)s[i];
}

// convert + transpose: src f32 [R, Cc]  ->  dst bf16 [Cc, R]
__global__ void cvt_t_kernel(const float* __restrict__ s, __bf16* __restrict__ d,
                             int R, int Cc) {
  size_t i = (size_t)blockIdx.x * blockDim.x + threadIdx.x;
  if (i >= (size_t)R * Cc) return;
  int r = (int)(i / Cc), c = (int)(i % Cc);
  d[(size_t)c * R + r] = (__bf16)s[i];
}

// token shift: xx = x_prev - x ; mixin = bf16(x + xx * time_maa_x);
// also spill x[:, -1] to output region 2.
__global__ void token_shift_kernel(const float* __restrict__ x,
                                   const float* __restrict__ sstate,
                                   const float* __restrict__ tmx,
                                   float* __restrict__ xx,
                                   __bf16* __restrict__ mixin,
                                   float* __restrict__ xlast) {
  size_t idx = (size_t)blockIdx.x * blockDim.x + threadIdx.x;
  if (idx >= (size_t)BT * C) return;
  int    c   = (int)(idx % C);
  size_t row = idx / C;
  int    t   = (int)(row % T);
  int    b   = (int)(row / T);
  float xv = x[idx];
  float xp = (t == 0) ? sstate[(size_t)b * C + c] : x[idx - C];
  float d  = xp - xv;
  xx[idx]    = d;
  mixin[idx] = (__bf16)(xv + d * tmx[c]);
  if (t == T - 1) xlast[(size_t)b * C + c] = xv;
}

// xxx = bf16(tanh(mixin[BT,2048] @ w1[2048,160])), w1 pre-transposed [160,2048]
__global__ __launch_bounds__(32)
void gemm_w1_tanh_kernel(const __bf16* __restrict__ A, const __bf16* __restrict__ W1t,
                         __bf16* __restrict__ xxx) {
  int row0 = blockIdx.x * 16, col0 = blockIdx.y * 16;
  v8f acc = {};
  for (int k0 = 0; k0 < C; k0 += 32) {
    v16bf a = load_a_frag(A, C, row0, k0);
    v16bf b = load_b_fragT(W1t, C, col0, k0);
    acc = wmma_bf16(a, b, acc);
  }
  int lane = threadIdx.x & 31;
  int n = lane & 15, mofs = (lane >> 4) * 8;
#pragma unroll
  for (int r = 0; r < 8; ++r)
    xxx[(size_t)(row0 + r + mofs) * 160 + (col0 + n)] = (__bf16)fast_tanh(acc[r]);
}

// m_f = xxx[:, f*32:(f+1)*32] @ w2[f]  (K=32, single WMMA, w2 pre-transposed [5,C,32])
// then x5[f] = bf16(x + xx * (time_maa_f + m_f))
__global__ __launch_bounds__(32)
void mix5_kernel(const __bf16* __restrict__ xxx, const __bf16* __restrict__ w2t,
                 const float* __restrict__ x, const float* __restrict__ xx,
                 const float* __restrict__ maa_w, const float* __restrict__ maa_k,
                 const float* __restrict__ maa_v, const float* __restrict__ maa_r,
                 const float* __restrict__ maa_g, __bf16* __restrict__ x5) {
  int f = blockIdx.z;
  int row0 = blockIdx.x * 16, col0 = blockIdx.y * 16;
  v16bf a = load_a_frag(xxx, 160, row0, f * 32);
  v16bf b = load_b_fragT(w2t + (size_t)f * C * 32, 32, col0, 0);
  v8f acc = {};
  acc = wmma_bf16(a, b, acc);
  const float* maa = (f == 0) ? maa_w : (f == 1) ? maa_k : (f == 2) ? maa_v
                   : (f == 3) ? maa_r : maa_g;
  __bf16* dst = x5 + (size_t)f * BT * C;
  int lane = threadIdx.x & 31;
  int n = lane & 15, mofs = (lane >> 4) * 8;
  int col = col0 + n;
  float mval = maa[col];
#pragma unroll
  for (int r = 0; r < 8; ++r) {
    size_t idx = (size_t)(row0 + r + mofs) * C + col;
    dst[idx] = (__bf16)(x[idx] + xx[idx] * (mval + acc[r]));
  }
}

// big GEMM: out[BT,C] = A[BT,C](bf16) @ W^T (W bf16 row-major [C,C])
// wave tile 32x64 (8 WMMA per 6 fragment loads); block = 4 waves = 32x256 cols.
// mode 0: store f32 ; mode 1: silu then store f32 ; mode 2: plain f32 (final out)
__device__ inline void store_tile(float* out, int row0, int col0, v8f d, int mode) {
  int lane = threadIdx.x & 31;
  int n = lane & 15, mofs = (lane >> 4) * 8;
  int col = col0 + n;
#pragma unroll
  for (int r = 0; r < 8; ++r) {
    float v = d[r];
    if (mode == 1) v = v * (1.0f / (1.0f + __expf(-v)));   // silu
    out[(size_t)(row0 + r + mofs) * C + col] = v;
  }
}

__global__ __launch_bounds__(128)
void gemm_big_kernel(const __bf16* __restrict__ A, const __bf16* __restrict__ W,
                     float* __restrict__ out, int mode) {
  int wave = threadIdx.x >> 5;
  int row0 = blockIdx.x * 32;
  int col0 = blockIdx.y * 256 + wave * 64;
  v8f acc[2][4] = {};
  for (int k0 = 0; k0 < C; k0 += 32) {
    v16bf fa0 = load_a_frag(A, C, row0, k0);
    v16bf fa1 = load_a_frag(A, C, row0 + 16, k0);
    v16bf fb0 = load_b_fragT(W, C, col0, k0);
    v16bf fb1 = load_b_fragT(W, C, col0 + 16, k0);
    v16bf fb2 = load_b_fragT(W, C, col0 + 32, k0);
    v16bf fb3 = load_b_fragT(W, C, col0 + 48, k0);
    acc[0][0] = wmma_bf16(fa0, fb0, acc[0][0]);
    acc[0][1] = wmma_bf16(fa0, fb1, acc[0][1]);
    acc[0][2] = wmma_bf16(fa0, fb2, acc[0][2]);
    acc[0][3] = wmma_bf16(fa0, fb3, acc[0][3]);
    acc[1][0] = wmma_bf16(fa1, fb0, acc[1][0]);
    acc[1][1] = wmma_bf16(fa1, fb1, acc[1][1]);
    acc[1][2] = wmma_bf16(fa1, fb2, acc[1][2]);
    acc[1][3] = wmma_bf16(fa1, fb3, acc[1][3]);
  }
#pragma unroll
  for (int i = 0; i < 2; ++i)
#pragma unroll
    for (int j = 0; j < 4; ++j)
      store_tile(out, row0 + i * 16, col0 + j * 16, acc[i][j], mode);
}

// hbuf = bf16(tanh(xw[BT,C] @ dw1[C,64])), dw1 pre-transposed [64,C]
__global__ __launch_bounds__(32)
void decay1_kernel(const __bf16* __restrict__ xw, const __bf16* __restrict__ dw1t,
                   __bf16* __restrict__ hbuf) {
  int row0 = blockIdx.x * 16, col0 = blockIdx.y * 16;
  v8f acc = {};
  for (int k0 = 0; k0 < C; k0 += 32) {
    v16bf a = load_a_frag(xw, C, row0, k0);
    v16bf b = load_b_fragT(dw1t, C, col0, k0);
    acc = wmma_bf16(a, b, acc);
  }
  int lane = threadIdx.x & 31;
  int n = lane & 15, mofs = (lane >> 4) * 8;
#pragma unroll
  for (int r = 0; r < 8; ++r)
    hbuf[(size_t)(row0 + r + mofs) * 64 + (col0 + n)] = (__bf16)fast_tanh(acc[r]);
}

// ew = exp(-exp(time_decay + hbuf[BT,64] @ dw2[64,C])), dw2 pre-transposed [C,64]
__global__ __launch_bounds__(32)
void decay2_kernel(const __bf16* __restrict__ hbuf, const __bf16* __restrict__ dw2t,
                   const float* __restrict__ tdecay, float* __restrict__ ew) {
  int row0 = blockIdx.x * 16, col0 = blockIdx.y * 16;
  v8f acc = {};
#pragma unroll
  for (int k0 = 0; k0 < 64; k0 += 32) {
    v16bf a = load_a_frag(hbuf, 64, row0, k0);
    v16bf b = load_b_fragT(dw2t, 64, col0, k0);
    acc = wmma_bf16(a, b, acc);
  }
  int lane = threadIdx.x & 31;
  int n = lane & 15, mofs = (lane >> 4) * 8;
  int col = col0 + n;
  float td = tdecay[col];
#pragma unroll
  for (int r = 0; r < 8; ++r) {
    float w = td + acc[r];
    ew[(size_t)(row0 + r + mofs) * C + col] = __expf(-__expf(w));
  }
}

// WKV6 recurrence: one block per (b,h), 64 threads, thread j owns state column j.
__global__ __launch_bounds__(64)
void wkv_scan_kernel(const float* __restrict__ r, const float* __restrict__ k,
                     const float* __restrict__ v, const float* __restrict__ ew,
                     const float* __restrict__ u, const float* __restrict__ S0,
                     float* __restrict__ xo, float* __restrict__ Sout) {
  int bh = blockIdx.x;
  int b = bh / H, h = bh % H;
  int j = threadIdx.x;                 // 0..63 : column index
  __shared__ float rr[64], kk[64], ww[64], uu[64];
  uu[j] = u[h * 64 + j];
  float S[64];
#pragma unroll
  for (int i = 0; i < 64; ++i)
    S[i] = S0[((size_t)(b * H + h) * 64 + i) * 64 + j];
  size_t base = (size_t)b * T * C + (size_t)h * 64;
  for (int t = 0; t < T; ++t) {
    size_t off = base + (size_t)t * C;
    __syncthreads();                   // protect LDS vectors from prev step
    rr[j] = r[off + j];
    kk[j] = k[off + j];
    ww[j] = ew[off + j];
    __syncthreads();
    float vj = v[off + j];
    float y = 0.0f;
#pragma unroll
    for (int i = 0; i < 64; ++i) {
      float kv = kk[i] * vj;
      y += rr[i] * (S[i] + uu[i] * kv);
      S[i] = S[i] * ww[i] + kv;
    }
    xo[off + j] = y;
  }
#pragma unroll
  for (int i = 0; i < 64; ++i)
    Sout[((size_t)(b * H + h) * 64 + i) * 64 + j] = S[i];
}

// GroupNorm(H groups over N=64) + gate with g, emit bf16 for final GEMM.
// 256 threads = 8 warps; each warp handles one (row, head).
__global__ __launch_bounds__(256)
void gn_gate_kernel(const float* __restrict__ xo, const float* __restrict__ g,
                    const float* __restrict__ gnw, const float* __restrict__ gnb,
                    __bf16* __restrict__ gated) {
  int warp = threadIdx.x >> 5, lane = threadIdx.x & 31;
  int gidx = blockIdx.x * 8 + warp;          // row*H + h
  int row = gidx / H, h = gidx % H;
  size_t base = (size_t)row * C + (size_t)h * 64;
  float a  = xo[base + lane];
  float bb = xo[base + lane + 32];
  float s = a + bb, s2 = a * a + bb * bb;
#pragma unroll
  for (int m = 16; m >= 1; m >>= 1) {
    s  += __shfl_xor(s,  m, 32);
    s2 += __shfl_xor(s2, m, 32);
  }
  float mean = s * (1.0f / 64.0f);
  float var  = s2 * (1.0f / 64.0f) - mean * mean;
  float rstd = rsqrtf(var + GN_EPS);
  int c0 = h * 64 + lane, c1 = c0 + 32;
  float y0 = (a  - mean) * rstd * gnw[c0] + gnb[c0];
  float y1 = (bb - mean) * rstd * gnw[c1] + gnb[c1];
  gated[base + lane]      = (__bf16)(y0 * g[base + lane]);
  gated[base + lane + 32] = (__bf16)(y1 * g[base + lane + 32]);
}

// ---------------- host launch ----------------
extern "C" void kernel_launch(void* const* d_in, const int* in_sizes, int n_in,
                              void* d_out, int out_size, void* d_ws, size_t ws_size,
                              hipStream_t stream) {
  (void)in_sizes; (void)n_in; (void)out_size; (void)ws_size;
  const float* x      = (const float*)d_in[0];
  const float* sstate = (const float*)d_in[1];
  const float* wkv0   = (const float*)d_in[2];
  const float* tmx    = (const float*)d_in[3];
  const float* tmw    = (const float*)d_in[4];
  const float* tmk    = (const float*)d_in[5];
  const float* tmv    = (const float*)d_in[6];
  const float* tmr    = (const float*)d_in[7];
  const float* tmg    = (const float*)d_in[8];
  const float* w1     = (const float*)d_in[9];
  const float* w2     = (const float*)d_in[10];
  const float* tdec   = (const float*)d_in[11];
  const float* dw1    = (const float*)d_in[12];
  const float* dw2    = (const float*)d_in[13];
  const float* faaaa  = (const float*)d_in[14];
  const float* Wr     = (const float*)d_in[15];
  const float* Wk     = (const float*)d_in[16];
  const float* Wv     = (const float*)d_in[17];
  const float* Wg     = (const float*)d_in[18];
  const float* Wo     = (const float*)d_in[19];
  const float* gnw    = (const float*)d_in[20];
  const float* gnb    = (const float*)d_in[21];

  float* out_y     = (float*)d_out;                         // [B,T,C]
  float* out_xlast = out_y + (size_t)BT * C;                // [B,C]
  float* out_Sfin  = out_xlast + (size_t)B * C;             // [B,H,N,N]

  // bump allocator on workspace
  char* ws = (char*)d_ws;
  size_t o = 0;
  auto alloc = [&](size_t bytes) -> char* {
    char* p = ws + o;
    o = (o + bytes + 255) & ~(size_t)255;
    return p;
  };
  __bf16* wr16  = (__bf16*)alloc((size_t)2 * C * C);
  __bf16* wk16  = (__bf16*)alloc((size_t)2 * C * C);
  __bf16* wv16  = (__bf16*)alloc((size_t)2 * C * C);
  __bf16* wg16  = (__bf16*)alloc((size_t)2 * C * C);
  __bf16* wo16  = (__bf16*)alloc((size_t)2 * C * C);
  __bf16* w1t   = (__bf16*)alloc((size_t)2 * 160 * C);   // [160, C]
  __bf16* w2t   = (__bf16*)alloc((size_t)2 * 5 * C * 32);// [5, C, 32]
  __bf16* dw1t  = (__bf16*)alloc((size_t)2 * 64 * C);    // [64, C]
  __bf16* dw2t  = (__bf16*)alloc((size_t)2 * C * 64);    // [C, 64]
  float*  xxf   = (float*) alloc((size_t)4 * BT * C);
  __bf16* mix16 = (__bf16*)alloc((size_t)2 * BT * C);
  __bf16* xxx16 = (__bf16*)alloc((size_t)2 * BT * 160);
  __bf16* x5_16 = (__bf16*)alloc((size_t)2 * 5 * BT * C);
  float*  rf    = (float*) alloc((size_t)4 * BT * C);
  float*  kf    = (float*) alloc((size_t)4 * BT * C);
  float*  vf    = (float*) alloc((size_t)4 * BT * C);
  float*  ewf   = (float*) alloc((size_t)4 * BT * C);
  float*  gf    = (float*) alloc((size_t)4 * BT * C);
  float*  xof   = (float*) alloc((size_t)4 * BT * C);
  __bf16* h16   = (__bf16*)alloc((size_t)2 * BT * 64);
  __bf16* gat16 = (__bf16*)alloc((size_t)2 * BT * C);

  auto cvt = [&](const float* s, __bf16* d, size_t n) {
    cvt_bf16_kernel<<<(unsigned)((n + 255) / 256), 256, 0, stream>>>(s, d, n);
  };
  auto cvtT = [&](const float* s, __bf16* d, int R, int Cc) {
    size_t n = (size_t)R * Cc;
    cvt_t_kernel<<<(unsigned)((n + 255) / 256), 256, 0, stream>>>(s, d, R, Cc);
  };

  // 1) weights -> bf16 (big mats as-is; small mats transposed so every
  //    WMMA B-operand is K-contiguous -> single 32B fragment load)
  cvt(Wr, wr16, (size_t)C * C);
  cvt(Wk, wk16, (size_t)C * C);
  cvt(Wv, wv16, (size_t)C * C);
  cvt(Wg, wg16, (size_t)C * C);
  cvt(Wo, wo16, (size_t)C * C);
  cvtT(w1, w1t, C, 160);                       // [C,160] -> [160,C]
  for (int f = 0; f < 5; ++f)                  // [32,C]  -> [C,32] per head-mix
    cvtT(w2 + (size_t)f * 32 * C, w2t + (size_t)f * C * 32, 32, C);
  cvtT(dw1, dw1t, C, 64);                      // [C,64]  -> [64,C]
  cvtT(dw2, dw2t, 64, C);                      // [64,C]  -> [C,64]

  // 2) token shift (+ x[:, -1] spill)
  {
    size_t n = (size_t)BT * C;
    token_shift_kernel<<<(unsigned)((n + 255) / 256), 256, 0, stream>>>(
        x, sstate, tmx, xxf, mix16, out_xlast);
  }

  // 3) xxx = tanh(mixin @ w1)   [BT x 160]
  gemm_w1_tanh_kernel<<<dim3(BT / 16, 160 / 16), 32, 0, stream>>>(mix16, w1t, xxx16);

  // 4) 5-way LoRA mix -> xw,xk,xv,xr,xg (bf16)
  mix5_kernel<<<dim3(BT / 16, C / 16, 5), 32, 0, stream>>>(
      xxx16, w2t, x, xxf, tmw, tmk, tmv, tmr, tmg, x5_16);

  // 5) big projections (x5 order: w,k,v,r,g)
  dim3 gg(BT / 32, C / 256);
  gemm_big_kernel<<<gg, 128, 0, stream>>>(x5_16 + (size_t)3 * BT * C, wr16, rf, 0);
  gemm_big_kernel<<<gg, 128, 0, stream>>>(x5_16 + (size_t)1 * BT * C, wk16, kf, 0);
  gemm_big_kernel<<<gg, 128, 0, stream>>>(x5_16 + (size_t)2 * BT * C, wv16, vf, 0);
  gemm_big_kernel<<<gg, 128, 0, stream>>>(x5_16 + (size_t)4 * BT * C, wg16, gf, 1); // silu

  // 6) decay path: ew = exp(-exp(time_decay + tanh(xw@dw1)@dw2))
  decay1_kernel<<<dim3(BT / 16, 64 / 16), 32, 0, stream>>>(
      x5_16 + (size_t)0 * BT * C, dw1t, h16);
  decay2_kernel<<<dim3(BT / 16, C / 16), 32, 0, stream>>>(h16, dw2t, tdec, ewf);

  // 7) WKV6 linear recurrence (sequential over T, parallel over B*H)
  wkv_scan_kernel<<<B * H, 64, 0, stream>>>(rf, kf, vf, ewf, faaaa, wkv0, xof, out_Sfin);

  // 8) GroupNorm + gate -> bf16
  gn_gate_kernel<<<(BT * H) / 8, 256, 0, stream>>>(xof, gf, gnw, gnb, gat16);

  // 9) out = gated @ Wo^T  -> d_out
  gemm_big_kernel<<<gg, 128, 0, stream>>>(gat16, wo16, out_y, 2);
}
